// DynamicConv2d_84018150245003
// MI455X (gfx1250) — matrix operand order repaired and checked by reading, not compiled, not run on hardware
//
#include <hip/hip_runtime.h>

// ---------------------------------------------------------------------------
// DynamicConv2d on gfx1250 (MI455X), wave32 + WMMA bf16 + async global->LDS.
//
// out[b] = conv2d(x[b], W_eff[b]),  W_eff[b] = sum_k aw[b,k]*conv_w[k]
// (fold the K=4 expert mix into the weights: 59 -> 14.8 GFLOP).
//
// Implicit GEMM per batch: (O=64) x (K=576) x (P=12544), bf16 in / f32 acc,
// K axis ordered tap-major (kk = tap*64 + ci) so each 32-wide K slab has a
// single (tap,hh,ww) -> gather addresses are base + ci*PIX (immediate offsets).
// ---------------------------------------------------------------------------

typedef __attribute__((ext_vector_type(16))) __bf16 v16bf;
typedef __attribute__((ext_vector_type(8)))  float  v8f;
typedef int vi4 __attribute__((__vector_size__(16)));

union FragBF { v16bf bf; uint4 u[2]; };

__device__ __forceinline__ unsigned short f2bf(float f) {
    union { float f; unsigned int u; } v; v.f = f;
    unsigned int r = v.u + 0x7FFFu + ((v.u >> 16) & 1u);  // RNE-ish
    return (unsigned short)(r >> 16);
}

#define C_IN    64
#define H_IMG   112
#define W_IMG   112
#define PIX     (H_IMG * W_IMG)          // 12544
#define KRED    (C_IN * 9)               // 576, kk = tap*64 + ci
#define KSTEP   32
#define NSTEPS  (KRED / KSTEP)           // 18
#define TILE_N  128
#define LDSROW  40                       // padded row (halves): 80B, bank-friendly

// ---- CDNA5 async global->LDS (guarded; falls back to a plain 16B copy) ----
__device__ __forceinline__ void async_copy16(const unsigned short* src,
                                             unsigned short* dst) {
#if defined(__AMDGCN__) && __has_builtin(__builtin_amdgcn_global_load_async_to_lds_b128)
    __builtin_amdgcn_global_load_async_to_lds_b128(
        (__attribute__((address_space(1))) vi4*)src,
        (__attribute__((address_space(3))) vi4*)dst, 0, 0);
#else
    *(uint4*)dst = *(const uint4*)src;
#endif
}

__device__ __forceinline__ void wait_async_copies() {
#if defined(__AMDGCN__) && __has_builtin(__builtin_amdgcn_global_load_async_to_lds_b128)
#if __has_builtin(__builtin_amdgcn_s_wait_asynccnt)
    __builtin_amdgcn_s_wait_asynccnt(0);
#else
    asm volatile("s_wait_asynccnt 0" ::: "memory");
#endif
#endif
}

// ---------------------------------------------------------------------------
// 1) Global average pool: pooled[b*64+c] = mean(x[b,c,:,:])
// ---------------------------------------------------------------------------
__global__ __launch_bounds__(256) void pool_kernel(const float* __restrict__ x,
                                                   float* __restrict__ pooled) {
    int bc = blockIdx.x;
    const float4* p = (const float4*)(x + (size_t)bc * PIX);
    float s = 0.f;
    for (int i = threadIdx.x; i < PIX / 4; i += 256) {
        float4 v = p[i];
        s += (v.x + v.y) + (v.z + v.w);
    }
    __shared__ float red[256];
    red[threadIdx.x] = s;
    __syncthreads();
    for (int off = 128; off > 0; off >>= 1) {
        if ((int)threadIdx.x < off) red[threadIdx.x] += red[threadIdx.x + off];
        __syncthreads();
    }
    if (threadIdx.x == 0) pooled[bc] = red[0] * (1.0f / (float)PIX);
}

// ---------------------------------------------------------------------------
// 2) Attention MLP + softmax: aw[b*4+k]
// ---------------------------------------------------------------------------
__global__ __launch_bounds__(32) void attn_kernel(const float* __restrict__ pooled,
                                                  const float* __restrict__ w1,   // (16,64)
                                                  const float* __restrict__ w2,   // (4,16)
                                                  float* __restrict__ aw) {
    int b = threadIdx.x;
    if (b >= 16) return;
    float h[16];
    for (int j = 0; j < 16; ++j) {
        float s = 0.f;
        for (int c = 0; c < C_IN; ++c) s += pooled[b * C_IN + c] * w1[j * C_IN + c];
        h[j] = fmaxf(s, 0.f);
    }
    float lg[4], m = -1e30f;
    for (int k = 0; k < 4; ++k) {
        float s = 0.f;
        for (int j = 0; j < 16; ++j) s += h[j] * w2[k * 16 + j];
        lg[k] = s;
        m = fmaxf(m, s);
    }
    float den = 0.f;
    for (int k = 0; k < 4; ++k) { lg[k] = __expf(lg[k] - m); den += lg[k]; }
    float inv = 1.0f / den;
    for (int k = 0; k < 4; ++k) aw[b * 4 + k] = lg[k] * inv;
}

// ---------------------------------------------------------------------------
// 3) Effective weights, bf16, tap-major K:
//    w_eff[b][o][tap*64+ci] = bf16(sum_k aw[b,k]*conv_w[k,o,ci,tap])
// ---------------------------------------------------------------------------
__global__ __launch_bounds__(256) void weff_kernel(const float* __restrict__ conv_w,
                                                   const float* __restrict__ aw,
                                                   unsigned short* __restrict__ w_eff) {
    int idx = blockIdx.x * 256 + threadIdx.x;            // < 16*64*576
    int b   = idx / (C_IN * KRED);
    int rem = idx - b * (C_IN * KRED);                   // o*576 + kk'
    int o   = rem / KRED;
    int kkp = rem - o * KRED;
    int tap = kkp >> 6;                                  // 0..8
    int ci  = kkp & 63;
    float s = 0.f;
    for (int k = 0; k < 4; ++k)
        s += aw[b * 4 + k] *
             conv_w[(((size_t)k * C_IN + o) * C_IN + ci) * 9 + tap];
    w_eff[idx] = f2bf(s);
}

// ---------------------------------------------------------------------------
// 4) Implicit-GEMM conv with WMMA bf16.
//    Block = 256 thr (8 waves). Output tile: 64 (O) x 128 (pixels).
// ---------------------------------------------------------------------------
__global__ __launch_bounds__(256) void dynconv_wmma_kernel(const float* __restrict__ x,
                                                           const unsigned short* __restrict__ w_eff,
                                                           float* __restrict__ out) {
    __shared__ __align__(16) unsigned short Alds[2][64 * LDSROW];      // [o][kkl]
    __shared__ __align__(16) unsigned short Blds[2][TILE_N * LDSROW];  // [n][kkl]

    const int tid   = threadIdx.x;
    const int lane  = tid & 31;
    const int wid   = tid >> 5;          // 0..7
    const int lm    = lane & 15;
    const int lh    = lane >> 4;         // 0/1
    const int b     = blockIdx.y;
    const int pbase = blockIdx.x * TILE_N;

    const float* xb = x + (size_t)b * C_IN * PIX;
    const unsigned short* wb = w_eff + (size_t)b * 64 * KRED;

    // im2col coords: thread owns pixel n_gather and kkl pair {kpair,kpair+1}+4j
    const int n_gather = tid & (TILE_N - 1);
    const int kpair    = (tid >> 7) * 2;     // 0 or 2
    const int p_g = pbase + n_gather;
    const int h_g = p_g / W_IMG;
    const int w_g = p_g - h_g * W_IMG;

    // A-staging: thread copies 8 halves (16B) of one weight row
    const int a_o   = tid >> 2;          // 0..63
    const int a_kkl = (tid & 3) * 8;     // 0,8,16,24

    v8f acc[4] = {};

#define STAGE_A(BUF, KK0)                                                          \
    async_copy16(wb + (size_t)a_o * KRED + (KK0) + a_kkl,                          \
                 &Alds[BUF][a_o * LDSROW + a_kkl])

#define STAGE_B(BUF, KK0)                                                          \
    do {                                                                           \
        const int tap = (KK0) >> 6;               /* constant within slab */       \
        const int hh  = h_g + tap / 3 - 1;                                         \
        const int ww  = w_g + tap % 3 - 1;                                         \
        const bool ok = ((unsigned)hh < (unsigned)H_IMG) &&                        \
                        ((unsigned)ww < (unsigned)W_IMG);                          \
        const float* src = xb + ((size_t)(((KK0) & 63) + kpair)) * PIX +           \
                           hh * W_IMG + ww;                                        \
        unsigned int* dst =                                                        \
            (unsigned int*)&Blds[BUF][n_gather * LDSROW + kpair];                  \
        _Pragma("unroll")                                                          \
        for (int j = 0; j < 8; ++j) {                                              \
            float v0 = 0.f, v1 = 0.f;                                              \
            if (ok) {                                                              \
                v0 = src[(size_t)(4 * j) * PIX];                                   \
                v1 = src[(size_t)(4 * j + 1) * PIX];                               \
            }                                                                      \
            dst[2 * j] = (unsigned)f2bf(v0) | ((unsigned)f2bf(v1) << 16);          \
        }                                                                          \
    } while (0)

    STAGE_A(0, 0);
    STAGE_B(0, 0);

    int buf = 0;
    for (int ks = 0; ks < NSTEPS; ++ks) {
        wait_async_copies();                // async A-tile of `buf` has landed
        __syncthreads();                    // publish staging; prev compute done
        if (ks + 1 < NSTEPS) {              // prefetch next K-slab (overlaps WMMA)
            STAGE_A(buf ^ 1, (ks + 1) * KSTEP);
            STAGE_B(buf ^ 1, (ks + 1) * KSTEP);
        }

        // B fragment: K-base lh*8, elements 0..7 -> K+0..7, 8..15 -> K+16..23
        FragBF bf_;
        const uint4* pb = (const uint4*)(&Blds[buf][(wid * 16 + lm) * LDSROW + lh * 8]);
        bf_.u[0] = pb[0];
        bf_.u[1] = pb[2];

#pragma unroll
        for (int mt = 0; mt < 4; ++mt) {
            // A fragment per ISA 16-bit A layout: K-base lh*8,
            // elements 0..7 -> K+0..7, elements 8..15 -> K+16..23
            FragBF af;
            const unsigned short* ap = &Alds[buf][(mt * 16 + lm) * LDSROW + lh * 8];
            af.u[0] = *(const uint4*)(ap);
            af.u[1] = *(const uint4*)(ap + 16);
            acc[mt] = __builtin_amdgcn_wmma_f32_16x16x32_bf16(
                false, af.bf, false, bf_.bf, (short)0, acc[mt], false, false);
        }
        buf ^= 1;
    }

    // Epilogue: C/D layout — VGPR r: lanes0-15 M=r, lanes16-31 M=8+r; N = lane%16
    float* ob = out + (size_t)b * 64 * PIX + pbase + wid * 16 + lm;
    const int rbase = lh * 8;
#pragma unroll
    for (int mt = 0; mt < 4; ++mt) {
#pragma unroll
        for (int r = 0; r < 8; ++r) {
            int o = mt * 16 + rbase + r;
            ob[(size_t)o * PIX] = acc[mt][r];
        }
    }
#undef STAGE_A
#undef STAGE_B
}

// ---------------------------------------------------------------------------
// Launcher. d_in order: x, conv_w, att_w1, att_w2.  out: (16,64,112,112) f32.
// ws layout: [0,4KB) pooled | [4KB,+256B) aw | [8KB, +1.13MB) w_eff bf16
// ---------------------------------------------------------------------------
extern "C" void kernel_launch(void* const* d_in, const int* in_sizes, int n_in,
                              void* d_out, int out_size, void* d_ws, size_t ws_size,
                              hipStream_t stream) {
    const float* x      = (const float*)d_in[0];
    const float* conv_w = (const float*)d_in[1];
    const float* att_w1 = (const float*)d_in[2];
    const float* att_w2 = (const float*)d_in[3];
    float* out = (float*)d_out;

    char* ws = (char*)d_ws;
    float*          pooled = (float*)(ws);                 // 1024 floats
    float*          aw     = (float*)(ws + 4096);          // 64 floats
    unsigned short* w_eff  = (unsigned short*)(ws + 8192); // 16*64*576 halves

    pool_kernel<<<16 * C_IN, 256, 0, stream>>>(x, pooled);
    attn_kernel<<<1, 32, 0, stream>>>(pooled, att_w1, att_w2, aw);
    weff_kernel<<<(16 * C_IN * KRED) / 256, 256, 0, stream>>>(conv_w, aw, w_eff);

    dim3 grid(PIX / TILE_N, 16);   // 98 x 16 blocks
    dynconv_wmma_kernel<<<grid, 256, 0, stream>>>(x, w_eff, out);
}
